// TransformerBlock_28381143892085
// MI455X (gfx1250) — matrix operand order repaired and checked by reading, not compiled, not run on hardware
//
#include <hip/hip_runtime.h>

// ---------------- types ----------------
typedef __bf16 bf16;
typedef __attribute__((ext_vector_type(16))) __bf16 v16bf;
typedef __attribute__((ext_vector_type(8)))  float  v8f;

#define WMMA_BF16(a, b, c) \
  __builtin_amdgcn_wmma_f32_16x16x32_bf16(false, (a), false, (b), (short)0, (c), false, false)

// Problem constants (from reference)
#define BB   8
#define TT   1024
#define EE   1024
#define HH   16
#define HD   64
#define HID  4096
#define MM   (BB * TT)        // 8192 rows
#define QKVP (3 * EE)         // 3072

// ---------------- async global->LDS (CDNA5) with safe fallback ----------------
#if defined(__AMDGCN__) && \
    __has_builtin(__builtin_amdgcn_global_load_async_to_lds_b128) && \
    __has_builtin(__builtin_amdgcn_s_wait_asynccnt)
#define USE_ASYNC_LDS 1
#else
#define USE_ASYNC_LDS 0
#endif

// Exact parameter types per hipcc diagnostic: gcc-vector int4, AS1 (src) / AS3 (dst)
typedef int v4i __attribute__((vector_size(4 * sizeof(int))));
typedef __attribute__((address_space(1))) v4i g_v4i;
typedef __attribute__((address_space(3))) v4i l_v4i;

// Copy 16 bytes (8 bf16) global -> LDS. Async on CDNA5 (ASYNCcnt), else uint4 round trip.
__device__ __forceinline__ void cp16(const bf16* __restrict__ g, bf16* __restrict__ l) {
#if USE_ASYNC_LDS
  // flat->AS casts via integers: global addr == flat addr; LDS offset == flat[31:0] (ISA 10.2)
  __builtin_amdgcn_global_load_async_to_lds_b128(
      (g_v4i*)(unsigned long long)(const void*)g,
      (l_v4i*)(unsigned)(unsigned long long)(void*)l, 0, 0);
#else
  *(uint4*)l = *(const uint4*)g;
#endif
}

__device__ __forceinline__ void wait_async_le4() {
#if USE_ASYNC_LDS
  __builtin_amdgcn_s_wait_asynccnt(4);
#endif
}
__device__ __forceinline__ void wait_async_0() {
#if USE_ASYNC_LDS
  __builtin_amdgcn_s_wait_asynccnt(0);
#endif
}

// ---------------- f32 [K,N] -> bf16 transposed [N,K] (weights) ----------------
__global__ __launch_bounds__(256) void k_cvt_t(const float* __restrict__ W,
                                               bf16* __restrict__ Wt, int K, int N) {
  int n0 = blockIdx.x * 32, k0 = blockIdx.y * 32;
  int tx = threadIdx.x & 31;
  int ty = threadIdx.x >> 5;   // 0..7
  __shared__ float tile[32][33];
#pragma unroll
  for (int i = 0; i < 4; ++i) {
    int k = ty + i * 8;
    tile[k][tx] = W[(size_t)(k0 + k) * N + n0 + tx];
  }
  __syncthreads();
#pragma unroll
  for (int i = 0; i < 4; ++i) {
    int n = ty + i * 8;
    Wt[(size_t)(n0 + n) * K + k0 + tx] = (bf16)tile[tx][n];
  }
}

// ---------------- LayerNorm: one row (E=1024) per block, 256 threads ----------------
__global__ __launch_bounds__(256) void k_layernorm(const float* __restrict__ x,
                                                   const float* __restrict__ g,
                                                   const float* __restrict__ beta,
                                                   bf16* __restrict__ out) {
  int row = blockIdx.x;
  int tid = threadIdx.x;
  const float* xr = x + (size_t)row * EE;
  float vals[4];
  float s = 0.f, ss = 0.f;
#pragma unroll
  for (int i = 0; i < 4; ++i) {
    float v = xr[tid + i * 256];
    vals[i] = v;
    s += v;
    ss += v * v;
  }
  __shared__ float rs[256], rss[256];
  rs[tid] = s; rss[tid] = ss;
  __syncthreads();
  for (int off = 128; off > 0; off >>= 1) {
    if (tid < off) { rs[tid] += rs[tid + off]; rss[tid] += rss[tid + off]; }
    __syncthreads();
  }
  float mu   = rs[0] * (1.0f / EE);
  float var  = rss[0] * (1.0f / EE) - mu * mu;
  float rstd = rsqrtf(var + 1e-5f);
  bf16* orow = out + (size_t)row * EE;
#pragma unroll
  for (int i = 0; i < 4; ++i) {
    int c = tid + i * 256;
    orow[c] = (bf16)((vals[i] - mu) * rstd * g[c] + beta[c]);
  }
}

// ---------------- GELU (tanh approx, matches jax.nn.gelu default) ----------------
__device__ __forceinline__ float gelu_f(float v) {
  float u = 0.7978845608028654f * (v + 0.044715f * v * v * v);
  return 0.5f * v * (1.0f + tanhf(u));
}

// ---------------- Tiled WMMA GEMM: C[M,N] = A[M,K] @ Wt[N,K]^T (+bias,+gelu,+resid) --
// 256 threads = 8 waves. Macro tile 128x128, KC=32, double-buffered LDS.
// Wave tile 64(M) x 32(N) = 8 accumulators.
__global__ __launch_bounds__(256) void k_gemm(const bf16* __restrict__ A,
                                              const bf16* __restrict__ Wt,  // [N,K]
                                              const float* __restrict__ bias,
                                              const float* __restrict__ resid,
                                              float* __restrict__ outF,
                                              bf16* __restrict__ outB,
                                              int M, int N, int K, int doGelu) {
  int tid  = threadIdx.x;
  int lane = tid & 31, wid = tid >> 5;
  int hi   = (lane >> 4) & 1, ln = lane & 15;
  int wm   = wid >> 2, wn = wid & 3;
  int mB   = blockIdx.y * 128, nB = blockIdx.x * 128;

  __shared__ bf16 Alds[2][128 * 32];   // [row][k]
  __shared__ bf16 Blds[2][128 * 32];   // [n][k]   (weights pre-transposed in global)

  v8f acc[4][2];
#pragma unroll
  for (int i = 0; i < 4; ++i)
#pragma unroll
    for (int j = 0; j < 2; ++j)
#pragma unroll
      for (int v = 0; v < 8; ++v) acc[i][j][v] = 0.0f;

  // Stage one 128x32 chunk of A and Wt into LDS buffer `buf`.
  // 512 granules of 8 bf16 (16B) per matrix; 2 per thread, contiguous 32B per thread.
  auto stage = [&](int kc, int buf) {
#pragma unroll
    for (int i = 0; i < 2; ++i) {
      int gidx = tid * 2 + i;
      int r = gidx >> 2, sg = (gidx & 3) * 8;
      cp16(A  + (size_t)(mB + r) * K + kc + sg, &Alds[buf][r * 32 + sg]);
      cp16(Wt + (size_t)(nB + r) * K + kc + sg, &Blds[buf][r * 32 + sg]);
    }
  };

  int nchunks = K >> 5;
  stage(0, 0);
  for (int c = 0; c < nchunks; ++c) {
    int cur = c & 1;
    bool hasNext = (c + 1) < nchunks;
    if (hasNext) {
      stage((c + 1) << 5, cur ^ 1);   // prefetch next chunk into other buffer
      wait_async_le4();               // oldest 4 (current chunk) complete
    } else {
      wait_async_0();
    }
    __syncthreads();

    v16bf af[4], bfr[2];
#pragma unroll
    for (int i = 0; i < 4; ++i) {
      const bf16* p = Alds[cur] + (wm * 64 + i * 16 + ln) * 32 + hi * 8;
#pragma unroll
      for (int h = 0; h < 8; ++h) { af[i][h] = p[h]; af[i][h + 8] = p[16 + h]; }
    }
#pragma unroll
    for (int j = 0; j < 2; ++j) {
      const bf16* p = Blds[cur] + (wn * 32 + j * 16 + ln) * 32 + hi * 8;
#pragma unroll
      for (int h = 0; h < 8; ++h) { bfr[j][h] = p[h]; bfr[j][h + 8] = p[16 + h]; }
    }
#pragma unroll
    for (int i = 0; i < 4; ++i)
#pragma unroll
      for (int j = 0; j < 2; ++j)
        acc[i][j] = WMMA_BF16(af[i], bfr[j], acc[i][j]);
    __syncthreads();   // all waves done reading `cur` before it is restaged
  }

  // Epilogue. C/D layout: row = vgpr + 8*hi, col = lane&15
#pragma unroll
  for (int i = 0; i < 4; ++i)
#pragma unroll
    for (int j = 0; j < 2; ++j) {
      int coln = nB + wn * 32 + j * 16 + ln;
#pragma unroll
      for (int v = 0; v < 8; ++v) {
        int rown = mB + wm * 64 + i * 16 + v + 8 * hi;
        float val = acc[i][j][v];
        if (bias)  val += bias[coln];
        if (doGelu) val = gelu_f(val);
        if (resid) val += resid[(size_t)rown * N + coln];
        if (outF)  outF[(size_t)rown * N + coln] = val;
        if (outB)  outB[(size_t)rown * N + coln] = (bf16)val;
      }
    }
}

// ---------------- Flash attention ----------------
// grid = (T/64, H, B), block = 128 (4 waves x 16 q-rows). Keys in chunks of 32.
__global__ __launch_bounds__(128) void k_attention(const bf16* __restrict__ qkv,
                                                   bf16* __restrict__ y) {
  int qt   = blockIdx.x;       // q tile (64 rows)
  int head = blockIdx.y;
  int b    = blockIdx.z;
  int tid  = threadIdx.x, lane = tid & 31, w = tid >> 5;
  int hi   = (lane >> 4) & 1, ln = lane & 15;

  size_t rowBase = (size_t)b * TT;
  int qcol = head * HD;
  int kcol = EE + head * HD;
  int vcol = 2 * EE + head * HD;
  int qrow0 = qt * 64 + w * 16;

  __shared__ bf16 Klds[32 * 64];      // [key][d]
  __shared__ bf16 Vt[64 * 32];        // [d][key]  (transposed)
  __shared__ bf16 Plds[4][16 * 32];   // per-wave P scratch [row][key]

  // Q fragments (A layout, 16x32 each, K-dim = head dim chunks 0-31 / 32-63)
  v16bf qf[2];
  {
    const bf16* qp = qkv + (rowBase + qrow0 + ln) * QKVP + qcol;
#pragma unroll
    for (int f = 0; f < 2; ++f)
#pragma unroll
      for (int h = 0; h < 16; ++h) {
        int k = (h & 7) + ((h >> 3) << 4) + hi * 8 + f * 32;
        qf[f][h] = qp[k];
      }
  }

  v8f O[4];
#pragma unroll
  for (int nt = 0; nt < 4; ++nt)
#pragma unroll
    for (int v = 0; v < 8; ++v) O[nt][v] = 0.0f;
  float mrow[8], lrow[8];
#pragma unroll
  for (int v = 0; v < 8; ++v) { mrow[v] = -1e30f; lrow[v] = 0.0f; }

  const float scale = 0.125f;  // 1/sqrt(64)

  for (int kc = 0; kc < TT; kc += 32) {
    __syncthreads();  // previous iteration done reading LDS
    // K chunk: row-major copy, async path. 256 granules of 16B; 2 per thread.
#pragma unroll
    for (int i = 0; i < 2; ++i) {
      int gidx = tid * 2 + i;
      int r = gidx >> 3, sg = (gidx & 7) * 8;
      cp16(qkv + (rowBase + kc + r) * QKVP + kcol + sg, &Klds[r * 64 + sg]);
    }
    // V chunk: transposed scatter (2B granularity, VGPR path)
    {
      int r = tid >> 2;            // 0..31 key
      int seg = (tid & 3) * 16;    // d segment
      const bf16* vp = qkv + (rowBase + kc + r) * QKVP + vcol + seg;
#pragma unroll
      for (int j = 0; j < 16; ++j) Vt[(seg + j) * 32 + r] = vp[j];
    }
    wait_async_0();
    __syncthreads();

    // S = Q @ K^T : two 16x16 key tiles, K-dim = 64 (2 wmma steps)
    v8f sacc[2];
#pragma unroll
    for (int t2 = 0; t2 < 2; ++t2) {
      v8f cc;
#pragma unroll
      for (int v = 0; v < 8; ++v) cc[v] = 0.0f;
#pragma unroll
      for (int ks = 0; ks < 2; ++ks) {
        v16bf bk;
        const bf16* kb = Klds + (t2 * 16 + ln) * 64 + ks * 32 + hi * 8;
#pragma unroll
        for (int h = 0; h < 8; ++h) { bk[h] = kb[h]; bk[h + 8] = kb[16 + h]; }
        cc = WMMA_BF16(qf[ks], bk, cc);
      }
      sacc[t2] = cc;
    }

    // online softmax (row = vgpr v + 8*hi; 16-lane half-wave reductions)
#pragma unroll
    for (int v = 0; v < 8; ++v) {
      float s0 = sacc[0][v] * scale;
      float s1 = sacc[1][v] * scale;
      float rmx = fmaxf(s0, s1);
#pragma unroll
      for (int msk = 1; msk < 16; msk <<= 1) rmx = fmaxf(rmx, __shfl_xor(rmx, msk, 32));
      float mnew = fmaxf(mrow[v], rmx);
      float corr = __expf(mrow[v] - mnew);
      mrow[v] = mnew;
      float p0 = __expf(s0 - mnew);
      float p1 = __expf(s1 - mnew);
      float rsum = p0 + p1;
#pragma unroll
      for (int msk = 1; msk < 16; msk <<= 1) rsum += __shfl_xor(rsum, msk, 32);
      lrow[v] = lrow[v] * corr + rsum;
#pragma unroll
      for (int nt = 0; nt < 4; ++nt) O[nt][v] *= corr;
      Plds[w][(v + 8 * hi) * 32 + ln]      = (bf16)p0;
      Plds[w][(v + 8 * hi) * 32 + 16 + ln] = (bf16)p1;
    }
    __syncthreads();

    // P fragment (A layout 16x32) and O += P @ V
    v16bf pf;
    {
      const bf16* pp = &Plds[w][ln * 32 + hi * 8];
#pragma unroll
      for (int h = 0; h < 8; ++h) { pf[h] = pp[h]; pf[h + 8] = pp[16 + h]; }
    }
#pragma unroll
    for (int nt = 0; nt < 4; ++nt) {
      v16bf bv;
      const bf16* vb = Vt + (nt * 16 + ln) * 32 + hi * 8;
#pragma unroll
      for (int h = 0; h < 8; ++h) { bv[h] = vb[h]; bv[h + 8] = vb[16 + h]; }
      O[nt] = WMMA_BF16(pf, bv, O[nt]);
    }
  }

  // normalize and write y in [b, t, head*64 + d] layout (ready for proj GEMM)
#pragma unroll
  for (int nt = 0; nt < 4; ++nt)
#pragma unroll
    for (int v = 0; v < 8; ++v) {
      float val = O[nt][v] / lrow[v];
      int trow = qrow0 + v + 8 * hi;
      int col  = head * HD + nt * 16 + ln;
      y[(rowBase + trow) * EE + col] = (bf16)val;
    }
}

// ---------------- launcher ----------------
extern "C" void kernel_launch(void* const* d_in, const int* in_sizes, int n_in,
                              void* d_out, int out_size, void* d_ws, size_t ws_size,
                              hipStream_t stream) {
  (void)in_sizes; (void)n_in; (void)out_size; (void)ws_size;

  const float* x     = (const float*)d_in[0];
  const float* ln1_g = (const float*)d_in[1];
  const float* ln1_b = (const float*)d_in[2];
  const float* Wqkv  = (const float*)d_in[3];
  const float* bqkv  = (const float*)d_in[4];
  const float* Wproj = (const float*)d_in[5];
  const float* bproj = (const float*)d_in[6];
  const float* ln2_g = (const float*)d_in[7];
  const float* ln2_b = (const float*)d_in[8];
  const float* Wfc   = (const float*)d_in[9];
  const float* bfc   = (const float*)d_in[10];
  const float* Wout  = (const float*)d_in[11];
  const float* bout  = (const float*)d_in[12];

  char* ws = (char*)d_ws;
  size_t off = 0;
  auto carve = [&](size_t bytes) -> char* {
    char* p = ws + off;
    off += (bytes + 255) & ~(size_t)255;
    return p;
  };
  bf16*  wqkv_t  = (bf16*)carve((size_t)EE * 3 * EE * 2);   // [3E, E]   (N,K)
  bf16*  wproj_t = (bf16*)carve((size_t)EE * EE * 2);       // [E, E]
  bf16*  wfc_t   = (bf16*)carve((size_t)EE * HID * 2);      // [HID, E]
  bf16*  wout_t  = (bf16*)carve((size_t)HID * EE * 2);      // [E, HID]
  bf16*  xn1     = (bf16*)carve((size_t)MM * EE * 2);
  bf16*  qkv     = (bf16*)carve((size_t)MM * QKVP * 2);
  bf16*  yatt    = (bf16*)carve((size_t)MM * EE * 2);
  float* x1      = (float*)carve((size_t)MM * EE * 4);
  bf16*  h2      = (bf16*)carve((size_t)MM * EE * 2);
  bf16*  act     = (bf16*)carve((size_t)MM * HID * 2);

  // 1) weights -> bf16, transposed to [N,K] for contiguous-K LDS staging
  k_cvt_t<<<dim3(QKVP / 32, EE / 32), 256, 0, stream>>>(Wqkv,  wqkv_t,  EE, QKVP);
  k_cvt_t<<<dim3(EE / 32, EE / 32),   256, 0, stream>>>(Wproj, wproj_t, EE, EE);
  k_cvt_t<<<dim3(HID / 32, EE / 32),  256, 0, stream>>>(Wfc,   wfc_t,   EE, HID);
  k_cvt_t<<<dim3(EE / 32, HID / 32),  256, 0, stream>>>(Wout,  wout_t,  HID, EE);

  // 2) LN1
  k_layernorm<<<MM, 256, 0, stream>>>(x, ln1_g, ln1_b, xn1);

  // 3) QKV GEMM: [8192,1024] @ [1024,3072] -> bf16 qkv (+bias)
  k_gemm<<<dim3(QKVP / 128, MM / 128), 256, 0, stream>>>(
      xn1, wqkv_t, bqkv, nullptr, nullptr, qkv, MM, QKVP, EE, 0);

  // 4) attention -> yatt bf16 [B*T, E]
  k_attention<<<dim3(TT / 64, HH, BB), 128, 0, stream>>>(qkv, yatt);

  // 5) proj GEMM + residual x -> x1 f32
  k_gemm<<<dim3(EE / 128, MM / 128), 256, 0, stream>>>(
      yatt, wproj_t, bproj, x, x1, nullptr, MM, EE, EE, 0);

  // 6) LN2
  k_layernorm<<<MM, 256, 0, stream>>>(x1, ln2_g, ln2_b, h2);

  // 7) FC GEMM + bias + GELU -> act bf16 [8192, 4096]
  k_gemm<<<dim3(HID / 128, MM / 128), 256, 0, stream>>>(
      h2, wfc_t, bfc, nullptr, nullptr, act, MM, HID, EE, 1);

  // 8) Out GEMM + bias + residual x1 -> d_out f32
  k_gemm<<<dim3(EE / 128, MM / 128), 256, 0, stream>>>(
      act, wout_t, bout, x1, (float*)d_out, nullptr, MM, EE, HID, 0);
}
